// SelfAttention_85246510891565
// MI455X (gfx1250) — matrix operand order repaired and checked by reading, not compiled, not run on hardware
//
#include <hip/hip_runtime.h>
#include <math.h>

// ---------------------------------------------------------------------------
// MI455X (gfx1250) GQA self-attention, bf16 WMMA pipeline.
// B=2 S=2048 D=2048 H=32 KVH=8 HD=64 G=4
// Compute-bound (~223 GF vs ~110MB traffic) -> bf16 WMMA everywhere,
// fp32 accumulate; async global->LDS staging for GEMM tiles.
// ---------------------------------------------------------------------------

typedef __bf16 bf16_t;
typedef __attribute__((ext_vector_type(16))) __bf16 v16bf;
typedef __attribute__((ext_vector_type(8)))  __bf16 v8bf;
typedef __attribute__((ext_vector_type(8)))  float  v8f;
typedef __attribute__((ext_vector_type(4)))  int    v4i;

#define B_  2
#define S_  2048
#define D_  2048
#define H_  32
#define KVH_ 8
#define HD_ 64
#define G_  4

// ----- CDNA5 async global->LDS copy (ASYNCcnt-tracked), with fallback ------
#if __has_builtin(__builtin_amdgcn_global_load_async_to_lds_b128)
#define HAVE_ASYNC_LDS 1
#else
#define HAVE_ASYNC_LDS 0
#endif

#if HAVE_ASYNC_LDS
typedef __attribute__((address_space(1))) v4i as1_v4i;
typedef __attribute__((address_space(3))) v4i as3_v4i;
#endif

static __device__ __forceinline__ void cp_async16(bf16_t* lds, const bf16_t* g) {
#if HAVE_ASYNC_LDS
  __builtin_amdgcn_global_load_async_to_lds_b128(
      (as1_v4i*)(g), (as3_v4i*)(lds), 0, 0);
#else
  *(v8bf*)lds = *(const v8bf*)g;
#endif
}
static __device__ __forceinline__ void async_wait_all() {
#if HAVE_ASYNC_LDS
#if __has_builtin(__builtin_amdgcn_s_wait_asynccnt)
  __builtin_amdgcn_s_wait_asynccnt(0);
#else
  asm volatile("s_wait_asynccnt 0" ::: "memory");
#endif
#endif
}

union FragU { v16bf v; v8bf h[2]; };

// B-fragment: 16 contiguous bf16 along k
static __device__ __forceinline__ v16bf ld_frag_b(const bf16_t* p) {
  FragU u;
  u.h[0] = *(const v8bf*)(p);
  u.h[1] = *(const v8bf*)(p + 8);
  return u.v;
}
// A-fragment: 16 contiguous bf16 along k starting at k=2*kh8 (=16*half)
static __device__ __forceinline__ v16bf ld_frag_a(const bf16_t* row, int kh8) {
  FragU u;
  u.h[0] = *(const v8bf*)(row + 2 * kh8);
  u.h[1] = *(const v8bf*)(row + 2 * kh8 + 8);
  return u.v;
}

// ---------------------------------------------------------------------------
// fp32 -> bf16 conversion (grid-stride)
// ---------------------------------------------------------------------------
__global__ void cvt_f32_bf16(const float* __restrict__ in, bf16_t* __restrict__ out, size_t n) {
  size_t i = (size_t)blockIdx.x * blockDim.x + threadIdx.x;
  size_t stride = (size_t)gridDim.x * blockDim.x;
  for (; i < n; i += stride) out[i] = (bf16_t)in[i];
}

// ---------------------------------------------------------------------------
// WMMA GEMM: C[m,n] = sum_k A[m,k]*W[n,k] + bias[n]
// A: MxK bf16 row-major, W: NxK bf16 row-major.
// OUTF32=1 -> fp32 out, OUTF32=0 -> bf16 out.
// Block: 256 threads (8 waves), tile 128x128x32. Wave -> 32x64 (2x4 wmma tiles).
// ---------------------------------------------------------------------------
#define BM 128
#define BN 128
#define BK 32
#define LDT 40   // lds row stride (elems): 32 + 8 pad, keeps 16B alignment

template <int OUTF32, int HAS_BIAS>
__global__ __launch_bounds__(256) void gemm_wmma_bf16(
    const bf16_t* __restrict__ A, const bf16_t* __restrict__ W,
    const float* __restrict__ bias, void* __restrict__ Cout,
    int M, int N, int K)
{
  __shared__ bf16_t As[BM][LDT];
  __shared__ bf16_t Ws[BN][LDT];

  const int bm = blockIdx.y * BM;
  const int bn = blockIdx.x * BN;
  const int t    = threadIdx.x;
  const int wave = t >> 5;
  const int lane = t & 31;
  const int wm = wave >> 1;        // 0..3  (row group of 32)
  const int wn = wave & 1;         // 0..1  (col group of 64)
  const int half = lane >> 4;
  const int l15  = lane & 15;

  const int lr = t >> 1;           // loader row 0..127
  const int lc = (t & 1) * 16;     // loader col 0 or 16

  v8f acc[2][4] = {};

  for (int k0 = 0; k0 < K; k0 += BK) {
    const bf16_t* ap = A + (size_t)(bm + lr) * K + k0 + lc;
    const bf16_t* wp = W + (size_t)(bn + lr) * K + k0 + lc;
    __builtin_prefetch(ap + BK, 0, 1);   // global_prefetch next k-tile
    __builtin_prefetch(wp + BK, 0, 1);
    cp_async16(&As[lr][lc],     ap);
    cp_async16(&As[lr][lc + 8], ap + 8);
    cp_async16(&Ws[lr][lc],     wp);
    cp_async16(&Ws[lr][lc + 8], wp + 8);
    async_wait_all();
    __syncthreads();

    v16bf af[2], bfr[4];
    const int kh8  = half * 8;
    const int kh16 = half * 16;
#pragma unroll
    for (int at = 0; at < 2; ++at)
      af[at] = ld_frag_a(&As[wm * 32 + at * 16 + l15][0], kh8);
#pragma unroll
    for (int nt = 0; nt < 4; ++nt)
      bfr[nt] = ld_frag_b(&Ws[wn * 64 + nt * 16 + l15][kh16]);

#pragma unroll
    for (int at = 0; at < 2; ++at)
#pragma unroll
      for (int nt = 0; nt < 4; ++nt)
        acc[at][nt] = __builtin_amdgcn_wmma_f32_16x16x32_bf16(
            false, af[at], false, bfr[nt], (short)0, acc[at][nt], false, false);
    __syncthreads();
  }

  // Epilogue: C layout VGPR i -> row i + 8*half, lane&15 -> col
#pragma unroll
  for (int at = 0; at < 2; ++at)
#pragma unroll
    for (int nt = 0; nt < 4; ++nt) {
      const int col = bn + wn * 64 + nt * 16 + l15;
      const float bb = HAS_BIAS ? bias[col] : 0.0f;
      const int row0 = bm + wm * 32 + at * 16 + 8 * half;
#pragma unroll
      for (int i = 0; i < 8; ++i) {
        float v = acc[at][nt][i] + bb;
        size_t off = (size_t)(row0 + i) * N + col;
        if (OUTF32) ((float*)Cout)[off] = v;
        else        ((bf16_t*)Cout)[off] = (bf16_t)v;
      }
    }
}

// ---------------------------------------------------------------------------
// RoPE in-place on bf16 tensor (B,S,NH*64). One thread per (b,s,h,d<32) pair.
// ---------------------------------------------------------------------------
__global__ void rope_bf16(bf16_t* __restrict__ x, const int* __restrict__ pos,
                          int NH, size_t total) {
  size_t idx = (size_t)blockIdx.x * blockDim.x + threadIdx.x;
  if (idx >= total) return;
  int d = (int)(idx & 31);
  int h = (int)((idx >> 5) % NH);
  size_t bs = idx / ((size_t)32 * NH);          // b*S + s
  float p = (float)pos[bs];
  // inv_freq = 10000^(-2d/64)
  float freq = __expf(-((float)(2 * d) / 64.0f) * 9.210340371976184f);
  float ang = p * freq;
  float c, sn;
  __sincosf(ang, &c, &sn);
  size_t base = bs * ((size_t)NH * 64) + (size_t)h * 64;
  float x1 = (float)x[base + d];
  float x2 = (float)x[base + d + 32];
  x[base + d]      = (bf16_t)(x1 * c - x2 * sn);
  x[base + d + 32] = (bf16_t)(x2 * c + x1 * sn);
}

// ---------------------------------------------------------------------------
// Flash attention (causal, GQA). One wave per (b, head, 16-row q tile).
// Scores and P@V via bf16 WMMA; online softmax with half-wave shuffles.
// ---------------------------------------------------------------------------
__global__ __launch_bounds__(32) void flash_attn(
    const bf16_t* __restrict__ Q, const bf16_t* __restrict__ Kc,
    const bf16_t* __restrict__ Vc, bf16_t* __restrict__ Ctx)
{
  const int q0  = blockIdx.x * 16;
  const int h   = blockIdx.y;
  const int b   = blockIdx.z;
  const int kvh = h >> 2;                       // G = 4
  const int lane = threadIdx.x;
  const int half = lane >> 4;
  const int l15  = lane & 15;
  const float scale = 0.125f;                   // 1/sqrt(64)

  __shared__ bf16_t Klds[32][72];               // key rows, d contiguous (+pad)
  __shared__ bf16_t Vt[64][40];                 // V transposed: [d][key] (+pad)
  __shared__ bf16_t Plds[16][40];               // P tile 16x32 (+pad)

  // Q fragments (row = q0 + l15, k-dim = HD split into 2 slices of 32)
  v16bf aq[2];
  {
    const bf16_t* qp = Q + ((size_t)(b * S_ + q0 + l15) * (H_ * HD_)) + (size_t)h * HD_;
    const int kh8 = half * 8;
    aq[0] = ld_frag_a(qp,      kh8);
    aq[1] = ld_frag_a(qp + 32, kh8);
  }

  float mstat[8], lstat[8];
#pragma unroll
  for (int i = 0; i < 8; ++i) { mstat[i] = -1e30f; lstat[i] = 0.0f; }
  v8f co[4] = {};

  const int kend = q0 + 16;                     // exclusive causal key bound
  for (int kb = 0; kb < kend; kb += 32) {
    // --- stage K rows (coalesced) and V transposed ---
    {
      const bf16_t* kp = Kc + ((size_t)(b * S_ + kb + lane) * (KVH_ * HD_)) + (size_t)kvh * HD_;
      const bf16_t* vp = Vc + ((size_t)(b * S_ + kb + lane) * (KVH_ * HD_)) + (size_t)kvh * HD_;
#pragma unroll
      for (int j = 0; j < 8; ++j)
        *(v8bf*)&Klds[lane][j * 8] = *(const v8bf*)(kp + j * 8);
#pragma unroll
      for (int j = 0; j < 8; ++j) {
        v8bf vv = *(const v8bf*)(vp + j * 8);
#pragma unroll
        for (int e = 0; e < 8; ++e) Vt[j * 8 + e][lane] = vv[e];
      }
    }
    __syncthreads();

    // --- scores: S = Q @ K^T  (two 16x16 n-tiles over 32 keys) ---
    v8f sc[2];
#pragma unroll
    for (int nt = 0; nt < 2; ++nt) {
      v8f cc = {};
#pragma unroll
      for (int s = 0; s < 2; ++s) {
        v16bf bk = ld_frag_b(&Klds[nt * 16 + l15][s * 32 + half * 16]);
        cc = __builtin_amdgcn_wmma_f32_16x16x32_bf16(false, aq[s], false, bk,
                                                     (short)0, cc, false, false);
      }
      sc[nt] = cc;
    }

    // --- online softmax (row = i + 8*half across VGPR slots) ---
    float corr[8];
#pragma unroll
    for (int i = 0; i < 8; ++i) {
      int qrow = q0 + i + 8 * half;
      float s0 = sc[0][i] * scale;
      float s1 = sc[1][i] * scale;
      if (kb + l15 > qrow)      s0 = -1e30f;
      if (kb + 16 + l15 > qrow) s1 = -1e30f;
      float vmax = fmaxf(s0, s1);
#pragma unroll
      for (int off = 1; off < 16; off <<= 1)
        vmax = fmaxf(vmax, __shfl_xor(vmax, off, 32));
      float mn = fmaxf(mstat[i], vmax);
      float cr = __expf(mstat[i] - mn);
      float p0 = __expf(s0 - mn);
      float p1 = __expf(s1 - mn);
      float rs = p0 + p1;
#pragma unroll
      for (int off = 1; off < 16; off <<= 1)
        rs += __shfl_xor(rs, off, 32);
      lstat[i] = lstat[i] * cr + rs;
      mstat[i] = mn;
      corr[i]  = cr;
      Plds[i + 8 * half][l15]      = (bf16_t)p0;
      Plds[i + 8 * half][16 + l15] = (bf16_t)p1;
    }
#pragma unroll
    for (int vt = 0; vt < 4; ++vt)
#pragma unroll
      for (int i = 0; i < 8; ++i) co[vt][i] *= corr[i];

    __syncthreads();
    asm volatile("s_wait_dscnt 0" ::: "memory");

    // --- O += P @ V  (P as A-fragment via LDS round-trip, V^T as B) ---
    v16bf ap = ld_frag_a(&Plds[l15][0], half * 8);
#pragma unroll
    for (int vt = 0; vt < 4; ++vt) {
      v16bf bv = ld_frag_b(&Vt[vt * 16 + l15][half * 16]);
      co[vt] = __builtin_amdgcn_wmma_f32_16x16x32_bf16(false, ap, false, bv,
                                                       (short)0, co[vt], false, false);
    }
    __syncthreads();
  }

  // --- normalize and write ctx (B,S,H*HD) bf16 ---
#pragma unroll
  for (int i = 0; i < 8; ++i) {
    float inv = 1.0f / lstat[i];
    int row = q0 + i + 8 * half;
    size_t base = ((size_t)(b * S_ + row) * (H_ * HD_)) + (size_t)h * HD_;
#pragma unroll
    for (int vt = 0; vt < 4; ++vt)
      Ctx[base + vt * 16 + l15] = (bf16_t)(co[vt][i] * inv);
  }
}

// ---------------------------------------------------------------------------
// host-side launcher
// ---------------------------------------------------------------------------
extern "C" void kernel_launch(void* const* d_in, const int* in_sizes, int n_in,
                              void* d_out, int out_size, void* d_ws, size_t ws_size,
                              hipStream_t stream) {
  const float* X   = (const float*)d_in[0];
  const int*   pos = (const int*)d_in[1];
  const float* Wq  = (const float*)d_in[2];
  const float* bq  = (const float*)d_in[3];
  const float* Wk  = (const float*)d_in[4];
  const float* bk  = (const float*)d_in[5];
  const float* Wv  = (const float*)d_in[6];
  const float* bv  = (const float*)d_in[7];
  const float* Wo  = (const float*)d_in[8];
  float* out = (float*)d_out;

  const size_t MS = (size_t)B_ * S_;            // 4096 token rows

  // workspace layout (bf16 buffers, all sizes multiples of 256B)
  char* p = (char*)d_ws;
  bf16_t* Xb  = (bf16_t*)p; p += MS * D_ * 2;                 // 16 MB
  bf16_t* Wqb = (bf16_t*)p; p += (size_t)(H_ * HD_) * D_ * 2; //  8 MB
  bf16_t* Wkb = (bf16_t*)p; p += (size_t)(KVH_ * HD_) * D_ * 2;
  bf16_t* Wvb = (bf16_t*)p; p += (size_t)(KVH_ * HD_) * D_ * 2;
  bf16_t* Wob = (bf16_t*)p; p += (size_t)D_ * (H_ * HD_) * 2;
  bf16_t* Qb  = (bf16_t*)p; p += MS * (H_ * HD_) * 2;         // 16 MB
  bf16_t* Kb  = (bf16_t*)p; p += MS * (KVH_ * HD_) * 2;
  bf16_t* Vb  = (bf16_t*)p; p += MS * (KVH_ * HD_) * 2;
  bf16_t* Ctx = (bf16_t*)p; p += MS * (H_ * HD_) * 2;         // 16 MB

  // 1) fp32 -> bf16 conversions
  cvt_f32_bf16<<<8192, 256, 0, stream>>>(X,  Xb,  MS * D_);
  cvt_f32_bf16<<<4096, 256, 0, stream>>>(Wq, Wqb, (size_t)(H_ * HD_) * D_);
  cvt_f32_bf16<<<1024, 256, 0, stream>>>(Wk, Wkb, (size_t)(KVH_ * HD_) * D_);
  cvt_f32_bf16<<<1024, 256, 0, stream>>>(Wv, Wvb, (size_t)(KVH_ * HD_) * D_);
  cvt_f32_bf16<<<4096, 256, 0, stream>>>(Wo, Wob, (size_t)D_ * (H_ * HD_));

  // 2) QKV projections (bias fused, bf16 out)
  {
    dim3 gq((H_ * HD_) / BN, MS / BM);
    gemm_wmma_bf16<0, 1><<<gq, 256, 0, stream>>>(Xb, Wqb, bq, (void*)Qb,
                                                 (int)MS, H_ * HD_, D_);
    dim3 gk((KVH_ * HD_) / BN, MS / BM);
    gemm_wmma_bf16<0, 1><<<gk, 256, 0, stream>>>(Xb, Wkb, bk, (void*)Kb,
                                                 (int)MS, KVH_ * HD_, D_);
    gemm_wmma_bf16<0, 1><<<gk, 256, 0, stream>>>(Xb, Wvb, bv, (void*)Vb,
                                                 (int)MS, KVH_ * HD_, D_);
  }

  // 3) RoPE in-place on Q and K
  {
    size_t tq = MS * H_ * 32;
    size_t tk = MS * KVH_ * 32;
    rope_bf16<<<(unsigned)((tq + 255) / 256), 256, 0, stream>>>(Qb, pos, H_,  tq);
    rope_bf16<<<(unsigned)((tk + 255) / 256), 256, 0, stream>>>(Kb, pos, KVH_, tk);
  }

  // 4) causal GQA flash attention -> Ctx
  {
    dim3 g(S_ / 16, H_, B_);
    flash_attn<<<g, 32, 0, stream>>>(Qb, Kb, Vb, Ctx);
  }

  // 5) output projection -> fp32 d_out
  {
    dim3 go(D_ / BN, MS / BM);
    gemm_wmma_bf16<1, 0><<<go, 256, 0, stream>>>(Ctx, Wob, nullptr, (void*)out,
                                                 (int)MS, D_, H_ * HD_);
  }
}
